// HyboWaveNet_90460601188772
// MI455X (gfx1250) — matrix operand order repaired and checked by reading, not compiled, not run on hardware
//
#include <hip/hip_runtime.h>
#include <hip/hip_bf16.h>
#include <math.h>

typedef __attribute__((ext_vector_type(16))) _Float16 v16h;
typedef __attribute__((ext_vector_type(8)))  _Float16 v8h;
typedef __attribute__((ext_vector_type(8)))  float    v8f;

#define DF   128
#define EPSF 1e-5f
#define MAXN 0.95f
#define MT   4      // M-tiles (of 16 rows) per block; amortizes B-fragment loads

// ---------------------------------------------------------------------------
// Row-wise log0 of input x:  t = atanh(clip(||x||,<=1-eps)) * x / max(||x||,eps)
// ---------------------------------------------------------------------------
__device__ __forceinline__ float wave_sum(float v) {
#pragma unroll
  for (int off = 16; off > 0; off >>= 1) v += __shfl_xor(v, off, 32);
  return v;
}

__global__ __launch_bounds__(128)
void k_log0_rows(const float* __restrict__ x, _Float16* __restrict__ t, int n) {
  int row = blockIdx.x;
  if (row >= n) return;
  int tid = threadIdx.x;                 // 0..127
  float u = x[(long)row * DF + tid];
  float ss = wave_sum(u * u);
  __shared__ float parts[4];
  __shared__ float beta_s;
  if ((tid & 31) == 0) parts[tid >> 5] = ss;
  __syncthreads();
  if (tid == 0) {
    float r = sqrtf(parts[0] + parts[1] + parts[2] + parts[3]);
    r = fmaxf(r, EPSF);
    beta_s = atanhf(fminf(r, 1.0f - EPSF)) / r;
  }
  __syncthreads();
  t[(long)row * DF + tid] = (_Float16)(u * beta_s);
}

// ---------------------------------------------------------------------------
// Finalize aggregation: u = sum/max(cnt,1); t = log0(proj(exp0(u))) = beta*u
// ---------------------------------------------------------------------------
__global__ __launch_bounds__(128)
void k_finalize_rows(const float* __restrict__ sum, const float* __restrict__ cnt,
                     _Float16* __restrict__ t, int n) {
  int row = blockIdx.x;
  if (row >= n) return;
  int tid = threadIdx.x;
  float c = fmaxf(cnt[row], 1.0f);
  float u = sum[(long)row * DF + tid] / c;
  float ss = wave_sum(u * u);
  __shared__ float parts[4];
  __shared__ float beta_s;
  if ((tid & 31) == 0) parts[tid >> 5] = ss;
  __syncthreads();
  if (tid == 0) {
    float r = fmaxf(sqrtf(parts[0] + parts[1] + parts[2] + parts[3]), EPSF);
    float hn = fminf(tanhf(r), MAXN);            // exp0 norm, projection clamp
    beta_s = atanhf(fminf(hn, 1.0f - EPSF)) / r; // fused log0
  }
  __syncthreads();
  t[(long)row * DF + tid] = (_Float16)(u * beta_s);
}

// ---------------------------------------------------------------------------
// Zero fill
// ---------------------------------------------------------------------------
__global__ void k_zero(float* __restrict__ p, long n) {
  long i = (long)blockIdx.x * blockDim.x + threadIdx.x;
  long stride = (long)gridDim.x * blockDim.x;
  for (; i < n; i += stride) p[i] = 0.0f;
}

// ---------------------------------------------------------------------------
// Edge aggregation: sum[dst] += t[src] (f16 gather, f32 atomic accumulate),
// cnt[dst] += 1. Node table is 12.8MB << 192MB L2, so gathers + atomics
// stay on-chip.
// ---------------------------------------------------------------------------
__global__ __launch_bounds__(256)
void k_edge_agg(const int* __restrict__ ei, const _Float16* __restrict__ t,
                float* __restrict__ sum, float* __restrict__ cnt, int E) {
  long idx = (long)blockIdx.x * blockDim.x + threadIdx.x;
  long total = (long)E * 16;
  if (idx >= total) return;
  int e  = (int)(idx >> 4);
  int ch = (int)(idx & 15);
  int src = ei[e];
  int dst = ei[E + e];
  const v8h tv = *(const v8h*)(t + (long)src * DF + ch * 8);
  float* sd = sum + (long)dst * DF + ch * 8;
#pragma unroll
  for (int i = 0; i < 8; ++i) atomicAdd(&sd[i], (float)tv[i]);
  if (ch == 0) atomicAdd(&cnt[dst], 1.0f);
}

// ---------------------------------------------------------------------------
// Fused hyperbolic linear via WMMA:
//   u = A(f16 tangent) @ W^T + b      (v_wmma_f32_16x16x32_f16, K=128)
//   Hout = alpha*u  (= proj(exp0(u)), f32)
//   Tout = beta*u   (= log0(proj(exp0(u))), f16)
// Block: 256 threads = 8 waves. Each block computes MT*16 rows x 128 cols,
// reusing the register-resident B fragments (W) across MT row-tiles.
// All 4 A fragments are loaded before the WMMA chain so load latency
// overlaps WMMA execution (waits drain 0x6 -> 0x0).
// ---------------------------------------------------------------------------
__global__ __launch_bounds__(256)
void k_hyp_linear_wmma(const _Float16* __restrict__ A,   // [n x 128]
                       const float* __restrict__ W,      // [128 x 128] row-major (out,in)
                       const float* __restrict__ bias,   // [128]
                       float* __restrict__ Hout,         // [n x 128]
                       _Float16* __restrict__ Tout,      // [n x 128]
                       int n) {
  __shared__ float tile[16][132];   // u tile (padded vs bank conflicts)
  __shared__ float part[16][16];
  __shared__ float alpha_s[16];
  __shared__ float beta_s[16];

  const int tid  = threadIdx.x;
  const int wave = tid >> 5;          // 0..7
  const int lane = tid & 31;
  const int lm   = lane & 15;         // 0..15
  const int g    = lane >> 4;         // 0..1
  const int n_base = wave * 16;
  const int ncol   = n_base + lm;     // output column == row of W

  // --- B fragments (loaded once, reused for MT row tiles):
  //     B[k][n] = W[n][k]; contiguous k order used here (same as A):
  //     lane (n=lm, g), half i -> k = kc*32 + 16*g + i (16 contiguous f32)
  v16h bfrag[4];
#pragma unroll
  for (int kc = 0; kc < 4; ++kc) {
    const float* wp = W + (long)ncol * DF + kc * 32 + 16 * g;
#pragma unroll
    for (int i = 0; i < 16; ++i) bfrag[kc][i] = (_Float16)wp[i];
  }
  const float bn = bias[ncol];

  const int row = tid >> 4;           // epilogue mapping
  const int seg = tid & 15;

  for (int mt = 0; mt < MT; ++mt) {
    const int m_base = (blockIdx.x * MT + mt) * 16;

    // --- A fragments: lane (m=lm, g), halves 0..7 -> k = kc*32+16g+i,
    //     halves 8..15 -> k = kc*32+16g+8+i (two contiguous 16B runs).
    //     Load ALL fragments first so the 8 b128 loads issue as one clause
    //     and drain while the WMMA chain executes.
    const int mrow = min(m_base + lm, n - 1);
    const _Float16* ap = A + (long)mrow * DF;
    v16h afrag[4];
#pragma unroll
    for (int kc = 0; kc < 4; ++kc) {
      const _Float16* p0 = ap + kc * 32 + 16 * g;
      const _Float16* p1 = p0 + 8;
#pragma unroll
      for (int i = 0; i < 8; ++i) { afrag[kc][i] = p0[i]; afrag[kc][i + 8] = p1[i]; }
    }

    v8f c = {};
#pragma unroll
    for (int kc = 0; kc < 4; ++kc) {
      c = __builtin_amdgcn_wmma_f32_16x16x32_f16(
          /*neg_a=*/false, afrag[kc], /*neg_b=*/false, bfrag[kc],
          /*c_mod=*/(short)0, c, /*reuse_a=*/false, /*reuse_b=*/false);
    }

    // --- epilogue: u = c + bias into LDS (C layout: VGPR r -> row r+8g, col lm)
#pragma unroll
    for (int r = 0; r < 8; ++r) tile[r + 8 * g][n_base + lm] = c[r] + bn;
    __syncthreads();

    {
      float s = 0.0f;
#pragma unroll
      for (int i = 0; i < 8; ++i) { float v = tile[row][seg * 8 + i]; s += v * v; }
      part[row][seg] = s;
    }
    __syncthreads();
    if (tid < 16) {
      float r2 = 0.0f;
#pragma unroll
      for (int i = 0; i < 16; ++i) r2 += part[tid][i];
      float r = fmaxf(sqrtf(r2), EPSF);
      float hn = tanhf(r);
      float alpha = hn / r;                              // exp0 scale
      if (hn > MAXN) { alpha = MAXN / r; hn = MAXN; }    // projection
      alpha_s[tid] = alpha;
      beta_s[tid]  = atanhf(fminf(hn, 1.0f - EPSF)) / r; // fused log0 scale
    }
    __syncthreads();
    {
      const int mr = m_base + row;
      if (mr < n) {
        const float al = alpha_s[row];
        const float be = beta_s[row];
        float*     hp = Hout + (long)mr * DF + seg * 8;
        _Float16*  tp = Tout + (long)mr * DF + seg * 8;
#pragma unroll
        for (int i = 0; i < 8; ++i) {
          float v = tile[row][seg * 8 + i];
          hp[i] = v * al;
          tp[i] = (_Float16)(v * be);
        }
      }
    }
    __syncthreads();   // protect LDS tile before next sub-tile overwrites it
  }
}

// ---------------------------------------------------------------------------
// Host-side orchestration
// ---------------------------------------------------------------------------
extern "C" void kernel_launch(void* const* d_in, const int* in_sizes, int n_in,
                              void* d_out, int out_size, void* d_ws, size_t ws_size,
                              hipStream_t stream) {
  const float* x     = (const float*)d_in[0];
  const int*   ei    = (const int*)d_in[1];
  const float* msg_W = (const float*)d_in[2];
  const float* msg_b = (const float*)d_in[3];
  const float* upd_W = (const float*)d_in[4];
  const float* upd_b = (const float*)d_in[5];

  const int N = in_sizes[0] / DF;         // 50000
  const int E = in_sizes[1] / 2;          // 800000
  const int L = in_sizes[2] / (DF * DF);  // 2

  // Workspace carve-up (256B aligned). Hout of non-final linears aliases `sum`
  // (dead value; gets zeroed before the edge stage).
  char* ws = (char*)d_ws;
  auto al256 = [](size_t o) { return (o + 255) & ~(size_t)255; };
  size_t o = 0;
  _Float16* tA  = (_Float16*)(ws + o); o = al256(o + (size_t)N * DF * sizeof(_Float16));
  _Float16* tB  = (_Float16*)(ws + o); o = al256(o + (size_t)N * DF * sizeof(_Float16));
  float*    sum = (float*)(ws + o);    o = al256(o + (size_t)N * DF * sizeof(float));
  float*    cnt = (float*)(ws + o);    o = al256(o + (size_t)N * sizeof(float));
  float*    hdump = sum;

  const int rows_per_block = 16 * MT;
  const int gemm_blocks = (N + rows_per_block - 1) / rows_per_block;
  const long agg_threads = (long)E * 16;
  const int agg_blocks = (int)((agg_threads + 255) / 256);

  // initial tangent
  k_log0_rows<<<N, 128, 0, stream>>>(x, tA, N);

  _Float16* cur = tA;
  _Float16* nxt = tB;
  for (int l = 0; l < L; ++l) {
    // 1) hyperbolic message linear (WMMA); emits f16 tangent of h for gather
    k_hyp_linear_wmma<<<gemm_blocks, 256, 0, stream>>>(
        cur, msg_W + (long)l * DF * DF, msg_b + (long)l * DF, hdump, nxt, N);

    // 2) zero accumulators, then edge-wise tangent aggregation
    k_zero<<<2048, 256, 0, stream>>>(sum, (long)N * DF);
    k_zero<<<256, 256, 0, stream>>>(cnt, (long)N);
    k_edge_agg<<<agg_blocks, 256, 0, stream>>>(ei, nxt, sum, cnt, E);

    // 3) mean + proj(exp0) + log0 fused -> f16 tangent for update linear
    k_finalize_rows<<<N, 128, 0, stream>>>(sum, cnt, cur, N);

    // 4) hyperbolic update linear (WMMA); final layer writes h to d_out
    float* hout = (l == L - 1) ? (float*)d_out : hdump;
    k_hyp_linear_wmma<<<gemm_blocks, 256, 0, stream>>>(
        cur, upd_W + (long)l * DF * DF, upd_b + (long)l * DF, hout, nxt, N);

    // nxt now holds log0(h) == tangent input for next layer's message linear
    _Float16* t = cur; cur = nxt; nxt = t;
  }
}
